// GRACE_66537633349711
// MI455X (gfx1250) — matrix operand order repaired and checked by reading, compiled, not run on hardware
//
#include <hip/hip_runtime.h>
#include <hip/hip_bf16.h>
#include <math.h>

// ---------------------------------------------------------------------------
// GRACE forward loss on gfx1250 (MI455X).
// bf16 WMMA GEMM chain, async-LDS double-buffered staging, f32 reductions.
// ---------------------------------------------------------------------------

typedef __bf16 bf16_t;
typedef __bf16 v16bf __attribute__((ext_vector_type(16)));
typedef __bf16 v8bf  __attribute__((ext_vector_type(8)));
typedef float  v8f   __attribute__((ext_vector_type(8)));

#define GN 8192        // N rows
#define GF 2048        // features
#define GE 256         // embedding
#define GK 64          // clusters
#define EPS_T_C 1.0f

// ---------------------------------------------------------------------------
// CDNA5 async global->LDS copy (ASYNCcnt-tracked, no VGPR round trip)
// ---------------------------------------------------------------------------
__device__ __forceinline__ void async_copy_b128(void* lds_dst, const void* gsrc) {
  typedef __attribute__((address_space(3))) unsigned char as3_byte;
  unsigned loff = (unsigned)(unsigned long long)(as3_byte*)lds_dst;
  asm volatile("global_load_async_to_lds_b128 %0, %1, off"
               :: "v"(loff), "v"(gsrc)
               : "memory");
}
__device__ __forceinline__ void wait_async0() {
  asm volatile("s_wait_asynccnt 0x0" ::: "memory");
}
__device__ __forceinline__ void wait_async4() {
  asm volatile("s_wait_asynccnt 0x4" ::: "memory");
}

// ---------------------------------------------------------------------------
// f32 -> bf16 conversion (grid-stride, no transpose)
// ---------------------------------------------------------------------------
__global__ void k_cvt_bf16(const float* __restrict__ in, bf16_t* __restrict__ out,
                           long long n) {
  long long i = (long long)blockIdx.x * blockDim.x + threadIdx.x;
  long long stride = (long long)gridDim.x * blockDim.x;
  for (; i < n; i += stride) out[i] = (bf16_t)in[i];
}

// ---------------------------------------------------------------------------
// f32 -> bf16 conversion WITH transpose (tiled through LDS, coalesced both
// sides):  out[c*rows + r] = (bf16) in[r*cols + c]
// ---------------------------------------------------------------------------
__global__ void k_transcvt(const float* __restrict__ in, bf16_t* __restrict__ out,
                           int rows, int cols) {
  __shared__ bf16_t t[32][33];
  const int bx = blockIdx.x * 32;   // col base
  const int by = blockIdx.y * 32;   // row base
  const int x = threadIdx.x;        // 0..31
  const int y = threadIdx.y;        // 0..7
#pragma unroll
  for (int r = 0; r < 32; r += 8)
    t[y + r][x] = (bf16_t)in[(long long)(by + y + r) * cols + bx + x];
  __syncthreads();
#pragma unroll
  for (int r = 0; r < 32; r += 8)
    out[(long long)(bx + y + r) * rows + by + x] = t[x][y + r];
}

// ---------------------------------------------------------------------------
// zero a small accumulator region
// ---------------------------------------------------------------------------
__global__ void k_zero(float* __restrict__ p, int n) {
  int i = blockIdx.x * blockDim.x + threadIdx.x;
  if (i < n) p[i] = 0.0f;
}

// ---------------------------------------------------------------------------
// Tiled bf16 WMMA GEMM:  C[M,N] = A[M,K] * B[K,N] (+ bias[N])
// B is supplied N-MAJOR (BT[n*K + k]) so all LDS staging is b128-vectorized.
// Block tile 128x128, BK=32, 256 threads (8 waves), wave tile 64x32 (4x2 WMMA).
// Double-buffered async global->LDS staging (ping-pong on ASYNCcnt).
// Epilogue: Cf = f32 row-major; Cb = bf16, row-major (A-chaining) or
// N-major when transCb (B-chaining; contiguous 16B stores).
// Fragment layouts match CDNA5 WMMA lane layouts:
//   A frag (lane l): M = l%16, Ko = (l<16?0:8):  As[m][Ko..Ko+7], As[m][Ko+16..Ko+23]
//   B frag (lane l): N = l%16, Kb = (l<16?0:8):  Bs[n][Kb..Kb+7], Bs[n][Kb+16..Kb+23]
// ---------------------------------------------------------------------------
__device__ __forceinline__ void stage_tiles(const bf16_t* __restrict__ A,
                                            const bf16_t* __restrict__ BT,
                                            int lda, int K, int rowBase, int colBase,
                                            int k0, int tid,
                                            bf16_t (*As)[48], bf16_t (*Bs)[48]) {
  const int r = tid >> 1, h = tid & 1;
  const bf16_t* sa = A  + (long long)(rowBase + r) * lda + k0 + h * 16;
  const bf16_t* sb = BT + (long long)(colBase + r) * K   + k0 + h * 16;
  async_copy_b128(&As[r][h * 16],     sa);
  async_copy_b128(&As[r][h * 16 + 8], sa + 8);
  async_copy_b128(&Bs[r][h * 16],     sb);
  async_copy_b128(&Bs[r][h * 16 + 8], sb + 8);
}

__launch_bounds__(256)
__global__ void k_gemm_bf16(const bf16_t* __restrict__ A, const bf16_t* __restrict__ BT,
                            const float* __restrict__ bias,
                            float* __restrict__ Cf, bf16_t* __restrict__ Cb, int transCb,
                            int M, int Ndim, int K, int lda) {
  __shared__ __align__(16) bf16_t As[2][128][48];  // 48 = 32 + pad (96B rows)
  __shared__ __align__(16) bf16_t Bs[2][128][48];

  const int tid  = threadIdx.x;
  const int lane = tid & 31;
  const int wave = tid >> 5;
  const int rowBase = blockIdx.y * 128;
  const int colBase = blockIdx.x * 128;
  const int wm = (wave & 1) * 64;
  const int wn = (wave >> 1) * 32;
  const int l15 = lane & 15;
  const int hiHalf = lane >> 4;
  const int koA = hiHalf * 8;
  const int kbB = hiHalf * 8;

  v8f acc[4][2] = {};

  const int kTiles = K >> 5;

  // prime buffer 0
  stage_tiles(A, BT, lda, K, rowBase, colBase, 0, tid, As[0], Bs[0]);

  for (int kt = 0; kt < kTiles; ++kt) {
    const int cur = kt & 1;
    const bool hasNext = (kt + 1) < kTiles;

    if (hasNext)
      stage_tiles(A, BT, lda, K, rowBase, colBase, (kt + 1) << 5, tid,
                  As[cur ^ 1], Bs[cur ^ 1]);

    if (kt + 2 < kTiles) {  // prefetch K-tile kt+2 toward L2
      const int k0n = (kt + 2) << 5;
      __builtin_prefetch(A  + (long long)(rowBase + (tid >> 1)) * lda + k0n + (tid & 1) * 16, 0, 0);
      __builtin_prefetch(BT + (long long)(colBase + (tid >> 1)) * K   + k0n + (tid & 1) * 16, 0, 0);
    }

    // only this iteration's 4 async ops must have landed (in-order completion)
    if (hasNext) wait_async4(); else wait_async0();
    __syncthreads();

    // ---- WMMA compute on buffer `cur`: 4 M-tiles x 2 N-tiles ----
    v16bf afrag[4], bfrag[2];
#pragma unroll
    for (int mt = 0; mt < 4; ++mt) {
      const int m = wm + mt * 16 + l15;
      v8bf lo = *(const v8bf*)&As[cur][m][koA];
      v8bf hi = *(const v8bf*)&As[cur][m][koA + 16];
      v16bf f;
#pragma unroll
      for (int i = 0; i < 8; ++i) { f[i] = lo[i]; f[i + 8] = hi[i]; }
      afrag[mt] = f;
    }
#pragma unroll
    for (int nt = 0; nt < 2; ++nt) {
      const int n = wn + nt * 16 + l15;
      v8bf lo = *(const v8bf*)&Bs[cur][n][kbB];
      v8bf hi = *(const v8bf*)&Bs[cur][n][kbB + 16];
      v16bf f;
#pragma unroll
      for (int i = 0; i < 8; ++i) { f[i] = lo[i]; f[i + 8] = hi[i]; }
      bfrag[nt] = f;
    }

#pragma unroll
    for (int mt = 0; mt < 4; ++mt)
#pragma unroll
      for (int nt = 0; nt < 2; ++nt)
        acc[mt][nt] = __builtin_amdgcn_wmma_f32_16x16x32_bf16(
            false, afrag[mt], false, bfrag[nt], (short)0, acc[mt][nt], false, false);

    __syncthreads();
  }

  // ---- epilogue: C layout VGPR v -> M = v + (lane<16?0:8), N = lane%16 ----
  const int mOff = hiHalf * 8;
#pragma unroll
  for (int mt = 0; mt < 4; ++mt) {
#pragma unroll
    for (int nt = 0; nt < 2; ++nt) {
      const int gn = colBase + wn + nt * 16 + l15;
      const float bv = bias ? bias[gn] : 0.0f;
      const int gmBase = rowBase + wm + mt * 16 + mOff;
      if (Cf) {
#pragma unroll
        for (int v = 0; v < 8; ++v)
          Cf[(long long)(gmBase + v) * Ndim + gn] = acc[mt][nt][v] + bv;
      }
      if (Cb) {
        if (transCb) {  // N-major: 8 consecutive rows of one column -> 16B store
          v8bf pk;
#pragma unroll
          for (int v = 0; v < 8; ++v) pk[v] = (bf16_t)(acc[mt][nt][v] + bv);
          *(v8bf*)&Cb[(long long)gn * M + gmBase] = pk;
        } else {
#pragma unroll
          for (int v = 0; v < 8; ++v)
            Cb[(long long)(gmBase + v) * Ndim + gn] = (bf16_t)(acc[mt][nt][v] + bv);
        }
      }
    }
  }
}

// ---------------------------------------------------------------------------
// Q pass 1: per-row t-distribution kernel + row-normalize; accumulate f_k.
// One wave per row; lanes split E=256 as 8 each; shuffle reductions.
// ---------------------------------------------------------------------------
__global__ void k_qpass1(const float* __restrict__ Z, const float* __restrict__ mean,
                         float* __restrict__ Q, float* __restrict__ fk) {
  __shared__ float mm[GK];
  __shared__ float fks[GK];
  const int tid = threadIdx.x;

  if (tid < GK) {
    float s = 0.0f;
    const float* mr = mean + tid * GE;
    for (int e = 0; e < GE; ++e) { float v = mr[e]; s += v * v; }
    mm[tid] = s;
    fks[tid] = 0.0f;
  }
  __syncthreads();

  const int lane = tid & 31;
  const int wave = tid >> 5;
  const int row = blockIdx.x * 8 + wave;

  const float* zr = Z + (long long)row * GE + lane * 8;
  float z[8];
#pragma unroll
  for (int j = 0; j < 8; ++j) z[j] = zr[j];
  float zz = 0.0f;
#pragma unroll
  for (int j = 0; j < 8; ++j) zz += z[j] * z[j];
#pragma unroll
  for (int off = 16; off; off >>= 1) zz += __shfl_xor(zz, off, 32);

  float qv0 = 0.0f, qv1 = 0.0f, S = 0.0f;
  for (int k = 0; k < GK; ++k) {
    const float* mr = mean + k * GE + lane * 8;
    float d = 0.0f;
#pragma unroll
    for (int j = 0; j < 8; ++j) d += z[j] * mr[j];
#pragma unroll
    for (int off = 16; off; off >>= 1) d += __shfl_xor(d, off, 32);
    const float sq = zz - 2.0f * d + mm[k];
    const float qk = powf(sq / EPS_T_C + 1.0f, -(EPS_T_C + 1.0f) * 0.5f);
    S += qk;
    if ((k & 31) == lane) { if (k < 32) qv0 = qk; else qv1 = qk; }
  }
  const float invS = 1.0f / S;
  const float q0 = qv0 * invS, q1 = qv1 * invS;
  Q[(long long)row * GK + lane]      = q0;
  Q[(long long)row * GK + 32 + lane] = q1;
  atomicAdd(&fks[lane], q0);
  atomicAdd(&fks[32 + lane], q1);
  __syncthreads();
  if (tid < GK) atomicAdd(&fk[tid], fks[tid]);
}

// ---------------------------------------------------------------------------
// Q pass 2: P = rownorm(Q^2/f_k); loss_c partial = sum P*log(P/Q)
// ---------------------------------------------------------------------------
__global__ void k_qpass2(const float* __restrict__ Q, const float* __restrict__ fk,
                         float* __restrict__ lc_sum) {
  __shared__ float red[256];
  const int tid = threadIdx.x;
  const int row = blockIdx.x * blockDim.x + tid;
  float acc = 0.0f;
  if (row < GN) {
    const float* qr = Q + (long long)row * GK;
    float num[GK];
    float s = 0.0f;
#pragma unroll 8
    for (int k = 0; k < GK; ++k) {
      const float q = qr[k];
      const float n = q * q / fk[k];
      num[k] = n;
      s += n;
    }
    const float invs = 1.0f / s;
#pragma unroll 8
    for (int k = 0; k < GK; ++k) {
      const float p = num[k] * invs;
      acc += p * logf(p / qr[k]);
    }
  }
  red[tid] = acc;
  __syncthreads();
  for (int off = 128; off; off >>= 1) {
    if (tid < off) red[tid] += red[tid + off];
    __syncthreads();
  }
  if (tid == 0) atomicAdd(lc_sum, red[0]);
}

// ---------------------------------------------------------------------------
// BCE-with-logits partial sum over X_p vs X
// ---------------------------------------------------------------------------
__global__ void k_bce(const float* __restrict__ Xp, const float* __restrict__ X,
                      float* __restrict__ lr_sum, long long n) {
  __shared__ float red[256];
  const int tid = threadIdx.x;
  long long i = (long long)blockIdx.x * blockDim.x + tid;
  const long long stride = (long long)gridDim.x * blockDim.x;
  float acc = 0.0f;
  for (; i < n; i += stride) {
    const float xp = Xp[i], x = X[i];
    acc += fmaxf(xp, 0.0f) - xp * x + log1pf(expf(-fabsf(xp)));
  }
  red[tid] = acc;
  __syncthreads();
  for (int off = 128; off; off >>= 1) {
    if (tid < off) red[tid] += red[tid + off];
    __syncthreads();
  }
  if (tid == 0) atomicAdd(lr_sum, red[0]);
}

// ---------------------------------------------------------------------------
// Combine: out = mean(loss_r) + mean(loss_c)
// ---------------------------------------------------------------------------
__global__ void k_final(const float* __restrict__ lr, const float* __restrict__ lc,
                        float* __restrict__ out) {
  if (blockIdx.x == 0 && threadIdx.x == 0) {
    out[0] = lr[0] / ((float)GN * (float)GF) + lc[0] / ((float)GN * (float)GK);
  }
}

// ---------------------------------------------------------------------------
// Host driver
// ---------------------------------------------------------------------------
extern "C" void kernel_launch(void* const* d_in, const int* in_sizes, int n_in,
                              void* d_out, int out_size, void* d_ws, size_t ws_size,
                              hipStream_t stream) {
  (void)in_sizes; (void)n_in; (void)out_size; (void)ws_size;

  const float* X    = (const float*)d_in[0];
  const float* RW   = (const float*)d_in[1];
  const float* mean = (const float*)d_in[2];
  const float* ew0  = (const float*)d_in[3];
  const float* eb0  = (const float*)d_in[4];
  const float* ew1  = (const float*)d_in[5];
  const float* eb1  = (const float*)d_in[6];
  const float* ew2  = (const float*)d_in[7];
  const float* eb2  = (const float*)d_in[8];
  const float* dw0  = (const float*)d_in[9];
  const float* db0  = (const float*)d_in[10];
  const float* dw1  = (const float*)d_in[11];
  const float* db1  = (const float*)d_in[12];
  const float* dwo  = (const float*)d_in[13];
  const float* dbo  = (const float*)d_in[14];
  float* out = (float*)d_out;

  // ---- workspace carve-up (256B aligned) ----
  size_t off = 0;
  auto take = [&](size_t bytes) -> char* {
    char* p = (char*)d_ws + off;
    off += (bytes + 255) & ~(size_t)255;
    return p;
  };
  float*  accf  = (float*)take(256 * sizeof(float));          // [0..63]=fk, [64]=lr, [65]=lc
  bf16_t* Xb    = (bf16_t*)take((size_t)GN * GF * 2);
  bf16_t* RWTb  = (bf16_t*)take((size_t)GN * GN * 2);         // RW^T, row-major (M=8192,K=8192)
  bf16_t* w0T   = (bf16_t*)take((size_t)1024 * GF * 2);       // (N,K) = (1024,2048)
  bf16_t* w1T   = (bf16_t*)take((size_t)512 * 1024 * 2);
  bf16_t* w2T   = (bf16_t*)take((size_t)GE * 512 * 2);
  bf16_t* v0T   = (bf16_t*)take((size_t)512 * GE * 2);
  bf16_t* v1T   = (bf16_t*)take((size_t)1024 * 512 * 2);
  bf16_t* voT   = (bf16_t*)take((size_t)GF * 1024 * 2);
  bf16_t* H0b   = (bf16_t*)take((size_t)GN * 1024 * 2);
  bf16_t* H1b   = (bf16_t*)take((size_t)GN * 512 * 2);
  bf16_t* Z0bT  = (bf16_t*)take((size_t)GE * GN * 2);         // (N,K)=(256,8192) for RW GEMM
  float*  Zf    = (float*)take((size_t)GN * GE * 4);
  bf16_t* Zb    = (bf16_t*)take((size_t)GN * GE * 2);
  bf16_t* D0b   = (bf16_t*)take((size_t)GN * 512 * 2);
  bf16_t* D1b   = (bf16_t*)take((size_t)GN * 1024 * 2);
  float*  Xp    = (float*)take((size_t)GN * GF * 4);
  float*  Qbuf  = (float*)take((size_t)GN * GK * 4);

  float* fk = accf;
  float* lr = accf + 64;
  float* lc = accf + 65;

  // ---- zero accumulators ----
  k_zero<<<1, 128, 0, stream>>>(accf, 66);

  // ---- conversions ----
  {
    long long n = (long long)GN * GF;
    k_cvt_bf16<<<4096, 256, 0, stream>>>(X, Xb, n);
  }
  auto tcvt = [&](const float* src, bf16_t* dst, int rows, int cols) {
    dim3 g(cols / 32, rows / 32), b(32, 8);
    k_transcvt<<<g, b, 0, stream>>>(src, dst, rows, cols);
  };
  tcvt(RW,  RWTb, GN,   GN);    // RWT[m][k] = RW[k][m]
  tcvt(ew0, w0T,  GF,   1024);  // w0T[n][k]
  tcvt(ew1, w1T,  1024, 512);
  tcvt(ew2, w2T,  512,  GE);
  tcvt(dw0, v0T,  GE,   512);
  tcvt(dw1, v1T,  512,  1024);
  tcvt(dwo, voT,  1024, GF);

  // ---- GEMM chain (all B operands N-major) ----
  auto gemm = [&](const bf16_t* A, const bf16_t* BT, const float* bias,
                  float* Cf, bf16_t* Cb, int transCb,
                  int M, int Ndim, int K, int lda) {
    dim3 g(Ndim / 128, M / 128);
    k_gemm_bf16<<<g, 256, 0, stream>>>(A, BT, bias, Cf, Cb, transCb, M, Ndim, K, lda);
  };

  // encoder (affine only; detach has no forward effect)
  gemm(Xb,  w0T, eb0, nullptr, H0b,  0, GN, 1024, GF,   GF);
  gemm(H0b, w1T, eb1, nullptr, H1b,  0, GN, 512,  1024, 1024);
  gemm(H1b, w2T, eb2, nullptr, Z0bT, 1, GN, GE,   512,  512);   // Z0 stored N-major

  // Z = RW^T @ Z0
  gemm(RWTb, Z0bT, nullptr, Zf, Zb, 0, GN, GE, GN, GN);

  // decoder
  gemm(Zb,  v0T, db0, nullptr, D0b, 0, GN, 512,  GE,   GE);
  gemm(D0b, v1T, db1, nullptr, D1b, 0, GN, 1024, 512,  512);
  gemm(D1b, voT, dbo, Xp,      nullptr, 0, GN, GF, 1024, 1024);

  // ---- clustering losses ----
  k_qpass1<<<GN / 8, 256, 0, stream>>>(Zf, mean, Qbuf, fk);
  k_qpass2<<<GN / 256, 256, 0, stream>>>(Qbuf, fk, lc);

  // ---- reconstruction loss ----
  k_bce<<<4096, 256, 0, stream>>>(Xp, X, lr, (long long)GN * GF);

  // ---- combine ----
  k_final<<<1, 1, 0, stream>>>(lr, lc, out);
}
